// ContextMultiHeadAttn_51917564674144
// MI455X (gfx1250) — matrix operand order repaired and checked by reading, not compiled, not run on hardware
//
#include <hip/hip_runtime.h>

// ---------------- problem constants ----------------
#define QLEN   1024
#define KVLEN  1024
#define BATCH  8
#define DMODEL 1024
#define NHEAD  16
#define DHEAD  64

typedef __attribute__((ext_vector_type(16))) __bf16 v16bf;
typedef __attribute__((ext_vector_type(8)))  float  v8f;
typedef __attribute__((ext_vector_type(4)))  unsigned int u32x4;
typedef __attribute__((ext_vector_type(8)))  int i32x8;
typedef __attribute__((ext_vector_type(4)))  int i32x4;

// gcc-style vector to exactly match the builtin's declared parameter type
typedef int vi4 __attribute__((vector_size(16)));
typedef __attribute__((address_space(1))) vi4 vi4_as1;
typedef __attribute__((address_space(3))) vi4 vi4_as3;

#if __has_builtin(__builtin_amdgcn_global_load_async_to_lds_b128) && \
    __has_builtin(__builtin_amdgcn_s_wait_asynccnt)
#define HAVE_ASYNC_LDS 1
#else
#define HAVE_ASYNC_LDS 0
#endif

union Frag16 {            // one 16x32 (A) or 32x16 (B) bf16 fragment: 8 VGPRs
    v16bf v;
    uint4 q[2];
};

#if HAVE_ASYNC_LDS
// Per-lane async copy of 16 bytes global -> LDS.
// AS(1) pointers are 64-bit flat-global values; AS(3) pointers are the low 32
// bits of the flat address (ISA aperture rule), so the
// integer round-trips are numerically correct, and avoid illegal AS casts.
__device__ __forceinline__ void async_copy16(const void* gp, const void* lp)
{
    __builtin_amdgcn_global_load_async_to_lds_b128(
        (vi4_as1*)(unsigned long long)gp,
        (vi4_as3*)(unsigned int)(unsigned long long)lp, 0, 0);
}
#endif

// ---------------- LayerNorm: h -> hln (fp32 for residual) + bf16 for GEMM A ----
__global__ void __launch_bounds__(256) ln_kernel(
    const float* __restrict__ h, const float* __restrict__ gamma,
    const float* __restrict__ beta, float* __restrict__ hlnF,
    __bf16* __restrict__ hlnB)
{
    int row = blockIdx.x;                 // (i*BATCH + b), 8192 rows
    int t = threadIdx.x;                  // 256 threads, 4 elems each
    const float* x = h + (size_t)row * DMODEL;
    float v[4];
    float s = 0.f, s2 = 0.f;
#pragma unroll
    for (int e = 0; e < 4; ++e) {
        v[e] = x[t * 4 + e];
        s += v[e];
        s2 += v[e] * v[e];
    }
    __shared__ float red0[256];
    __shared__ float red1[256];
    red0[t] = s; red1[t] = s2;
    __syncthreads();
    for (int off = 128; off > 0; off >>= 1) {
        if (t < off) { red0[t] += red0[t + off]; red1[t] += red1[t + off]; }
        __syncthreads();
    }
    float mu  = red0[0] * (1.0f / DMODEL);
    float var = red1[0] * (1.0f / DMODEL) - mu * mu;
    float rstd = rsqrtf(var + 1e-5f);
#pragma unroll
    for (int e = 0; e < 4; ++e) {
        int d = t * 4 + e;
        float y = (v[e] - mu) * rstd * gamma[d] + beta[d];
        hlnF[(size_t)row * DMODEL + d] = y;
        hlnB[(size_t)row * DMODEL + d] = (__bf16)y;
    }
}

// ---------------- float -> bf16 elementwise ----------------
__global__ void f2b_kernel(const float* __restrict__ in, __bf16* __restrict__ out, int n)
{
    int i = blockIdx.x * blockDim.x + threadIdx.x;
    if (i < n) out[i] = (__bf16)in[i];
}

// ---------------- transpose + convert: W[K x N] fp32 -> WT[N x K] bf16 --------
__global__ void tconv_kernel(const float* __restrict__ in, __bf16* __restrict__ out,
                             int rows /*K*/, int cols /*N*/)
{
    int i = blockIdx.x * blockDim.x + threadIdx.x;
    if (i < rows * cols) {
        int k = i / cols, n = i - k * cols;
        out[(size_t)n * rows + k] = (__bf16)in[i];
    }
}

// ---------------- fragment helpers ----------------
__device__ __forceinline__ void load_b_frags(
    Frag16* fb, const __bf16* b0, const __bf16* b1, const __bf16* b2, const __bf16* b3,
    int k, int kbB)
{
    fb[0].q[0] = *(const uint4*)(b0 + k + kbB);
    fb[0].q[1] = *(const uint4*)(b0 + k + kbB + 16);
    fb[1].q[0] = *(const uint4*)(b1 + k + kbB);
    fb[1].q[1] = *(const uint4*)(b1 + k + kbB + 16);
    fb[2].q[0] = *(const uint4*)(b2 + k + kbB);
    fb[2].q[1] = *(const uint4*)(b2 + k + kbB + 16);
    fb[3].q[0] = *(const uint4*)(b3 + k + kbB);
    fb[3].q[1] = *(const uint4*)(b3 + k + kbB + 16);
}

__device__ __forceinline__ void load_a_frags_global(
    Frag16* fa, const __bf16* arow0, const __bf16* arow1, int k, int kbA)
{
    fa[0].q[0] = *(const uint4*)(arow0 + k + kbA);
    fa[0].q[1] = *(const uint4*)(arow0 + k + kbA + 16);
    fa[1].q[0] = *(const uint4*)(arow1 + k + kbA);
    fa[1].q[1] = *(const uint4*)(arow1 + k + kbA + 16);
}

// ---------------- WMMA GEMM: C[MxN] = A[MxK](bf16) * BT[NxK](bf16) ------------
// Block = 8 waves; all waves share one 32-row A tile (tiles_n % 8 == 0), which
// is staged 64-k-deep into LDS via async global->LDS copies (double buffered).
// Each wave computes a 32x64 tile: 2 A frags x 4 B frags per 32-k substep.
// mode 0: write Q bf16 scattered to [b][head][i][d]
// mode 1: write K bf16 to [b][head][j][d], V bf16 TRANSPOSED to [b][head][d][j]
// mode 2: out fp32 = resid + acc  (row-major [M x N])
__global__ void __launch_bounds__(256) gemm_kernel(
    const __bf16* __restrict__ A, const __bf16* __restrict__ BT,
    int M, int N, int K, int mode,
    const float* __restrict__ resid, float* __restrict__ outF,
    __bf16* __restrict__ outQ, __bf16* __restrict__ outK, __bf16* __restrict__ outV)
{
    int tid = threadIdx.x;
    int wave = (int)((blockIdx.x * blockDim.x + tid) >> 5);
    int lane = tid & 31;
    int tiles_n = N >> 6;
    int tm = wave / tiles_n;
    int tn = wave - tm * tiles_n;
    int m0 = tm << 5, n0 = tn << 6;

    int ln = lane & 15;
    int hi = (lane >> 4) & 1;
    int kbA = hi * 8;            // A fragment K-offset per lane half
    int kbB = hi * 8;            // B fragment K-offset per lane half

    v8f acc[2][4] = {};

    const __bf16* b0 = BT + (size_t)(n0 + 0 * 16 + ln) * K;
    const __bf16* b1 = BT + (size_t)(n0 + 1 * 16 + ln) * K;
    const __bf16* b2 = BT + (size_t)(n0 + 2 * 16 + ln) * K;
    const __bf16* b3 = BT + (size_t)(n0 + 3 * 16 + ln) * K;

#if HAVE_ASYNC_LDS
    __shared__ __bf16 aTile[2][32 * 64];    // 2 x 4 KiB double buffer
    // stage one 32x64 A tile: 256 threads x 16B
    int sr  = tid >> 3;                      // row 0..31
    int seg = tid & 7;                       // 16B segment 0..7
    const __bf16* agsrc = A + (size_t)(m0 + sr) * K + seg * 8;
    __bf16* aldst0 = &aTile[0][sr * 64 + seg * 8];
    __bf16* aldst1 = &aTile[1][sr * 64 + seg * 8];

    async_copy16(agsrc, aldst0);                       // stage k0 = 0
    for (int k0 = 0; k0 < K; k0 += 64) {
        int cur = (k0 >> 6) & 1;
        if (k0 + 64 < K) {
            async_copy16(agsrc + k0 + 64, cur ? aldst0 : aldst1);
            __builtin_amdgcn_s_wait_asynccnt(1);       // current tile landed
        } else {
            __builtin_amdgcn_s_wait_asynccnt(0);
        }
        __syncthreads();                               // all waves' copies done

        __builtin_prefetch(b0 + k0 + 512, 0, 1);
        __builtin_prefetch(b2 + k0 + 512, 0, 1);

        const __bf16* abuf = &aTile[cur][0];
#pragma unroll
        for (int ks = 0; ks < 2; ++ks) {               // two 32-k substeps
            Frag16 fb[4];
            load_b_frags(fb, b0, b1, b2, b3, k0 + ks * 32, kbB);
            Frag16 fa[2];
            const __bf16* lrow0 = abuf + (size_t)ln * 64 + ks * 32;
            const __bf16* lrow1 = abuf + (size_t)(16 + ln) * 64 + ks * 32;
            fa[0].q[0] = *(const uint4*)(lrow0 + kbA);
            fa[0].q[1] = *(const uint4*)(lrow0 + kbA + 16);
            fa[1].q[0] = *(const uint4*)(lrow1 + kbA);
            fa[1].q[1] = *(const uint4*)(lrow1 + kbA + 16);
#pragma unroll
            for (int s = 0; s < 2; ++s)
#pragma unroll
                for (int t = 0; t < 4; ++t)
                    acc[s][t] = __builtin_amdgcn_wmma_f32_16x16x32_bf16(
                        false, fa[s].v, false, fb[t].v, (short)0, acc[s][t], false, false);
        }
        __syncthreads();                               // buffer reuse guard
    }
#else
    // fallback: register double-buffered, direct global->VGPR A fragments
    const __bf16* arow0 = A + (size_t)(m0 + ln) * K;
    const __bf16* arow1 = A + (size_t)(m0 + 16 + ln) * K;
    Frag16 faA[2], faB[2], fbA[4], fbB[4];
    load_a_frags_global(faA, arow0, arow1, 0, kbA);
    load_b_frags(fbA, b0, b1, b2, b3, 0, kbB);
    for (int k0 = 0; k0 < K; k0 += 64) {
        load_a_frags_global(faB, arow0, arow1, k0 + 32, kbA);
        load_b_frags(fbB, b0, b1, b2, b3, k0 + 32, kbB);
#pragma unroll
        for (int s = 0; s < 2; ++s)
#pragma unroll
            for (int t = 0; t < 4; ++t)
                acc[s][t] = __builtin_amdgcn_wmma_f32_16x16x32_bf16(
                    false, faA[s].v, false, fbA[t].v, (short)0, acc[s][t], false, false);
        if (k0 + 64 < K) {
            load_a_frags_global(faA, arow0, arow1, k0 + 64, kbA);
            load_b_frags(fbA, b0, b1, b2, b3, k0 + 64, kbB);
        }
#pragma unroll
        for (int s = 0; s < 2; ++s)
#pragma unroll
            for (int t = 0; t < 4; ++t)
                acc[s][t] = __builtin_amdgcn_wmma_f32_16x16x32_bf16(
                    false, faB[s].v, false, fbB[t].v, (short)0, acc[s][t], false, false);
    }
#endif

#pragma unroll
    for (int s = 0; s < 2; ++s) {
        int mrow0 = m0 + s * 16 + hi * 8;
#pragma unroll
        for (int t = 0; t < 4; ++t) {
            int col = n0 + t * 16 + ln;
#pragma unroll
            for (int r = 0; r < 8; ++r) {
                int m = mrow0 + r;
                float val = acc[s][t][r];
                if (mode == 2) {
                    size_t idx = (size_t)m * N + col;
                    outF[idx] = resid[idx] + val;
                } else if (mode == 0) {
                    int i = m >> 3, b = m & 7;
                    int nh = col >> 6, d = col & 63;
                    outQ[(((size_t)b * NHEAD + nh) * QLEN + i) * DHEAD + d] = (__bf16)val;
                } else {
                    int j = m >> 3, b = m & 7;
                    if (col < NHEAD * DHEAD) {
                        int nh = col >> 6, d = col & 63;
                        outK[(((size_t)b * NHEAD + nh) * KVLEN + j) * DHEAD + d] = (__bf16)val;
                    } else {
                        int c2 = col - NHEAD * DHEAD;
                        int nh = c2 >> 6, d = c2 & 63;
                        // V transposed: [b][head][d][j]
                        outV[(((size_t)b * NHEAD + nh) * DHEAD + d) * KVLEN + j] = (__bf16)val;
                    }
                }
            }
        }
    }
}

// ---------------- Flash attention: per wave one (b, head, 16-row q tile) ------
// Q/K bf16 in [b][head][seq][DHEAD]; V bf16 TRANSPOSED in [b][head][DHEAD][j];
// out bf16 attn_vec [i][b][head*64+d].  64-key chunks per iteration.
__global__ void __launch_bounds__(256) attn_kernel(
    const __bf16* __restrict__ Q, const __bf16* __restrict__ Kb,
    const __bf16* __restrict__ Vt, const unsigned char* __restrict__ mask,
    __bf16* __restrict__ out)
{
    __shared__ __bf16 pLds[8][16 * 64];   // one 16x64 P tile per wave (16 KiB)
    int wslot = threadIdx.x >> 5;
    int lane = threadIdx.x & 31;
    int wave = (int)((blockIdx.x * blockDim.x + threadIdx.x) >> 5);
    int qt = wave & 63;                   // 64 q tiles of 16 rows
    int bn = wave >> 6;                   // 0..127 == b*16 + head
    int b = bn >> 4, nh = bn & 15;
    size_t headOff = (size_t)bn * KVLEN * DHEAD;

    int ln = lane & 15;
    int hi = (lane >> 4) & 1;
    int kbA = hi * 8;
    int kbB = hi * 8;

    // Q A-fragments for this wave's 16x64 tile (d split 0..31 / 32..63)
    const __bf16* qrow = Q + headOff + (size_t)(qt * 16 + ln) * DHEAD;
    Frag16 aq0, aq1;
    aq0.q[0] = *(const uint4*)(qrow + kbA);
    aq0.q[1] = *(const uint4*)(qrow + kbA + 16);
    aq1.q[0] = *(const uint4*)(qrow + 32 + kbA);
    aq1.q[1] = *(const uint4*)(qrow + 32 + kbA + 16);

    float mrow[8], lsum[8];
#pragma unroll
    for (int r = 0; r < 8; ++r) { mrow[r] = -3.0e38f; lsum[r] = 0.f; }
    v8f acc[4] = {};

    __bf16* pbuf = &pLds[wslot][0];

    for (int j0 = 0; j0 < KVLEN; j0 += 64) {
        float p[4][8];
        // ----- scores: four 16x16 tiles over d=64 (2 chained WMMAs each) ----
#pragma unroll
        for (int jt = 0; jt < 4; ++jt) {
            int jc = j0 + jt * 16 + ln;   // this lane's key column
            const __bf16* krow = Kb + headOff + (size_t)jc * DHEAD;
            Frag16 bk0, bk1;
            bk0.q[0] = *(const uint4*)(krow + kbB);
            bk0.q[1] = *(const uint4*)(krow + kbB + 16);
            bk1.q[0] = *(const uint4*)(krow + 32 + kbB);
            bk1.q[1] = *(const uint4*)(krow + 32 + kbB + 16);
            v8f s = {};
            s = __builtin_amdgcn_wmma_f32_16x16x32_bf16(
                false, aq0.v, false, bk0.v, (short)0, s, false, false);
            s = __builtin_amdgcn_wmma_f32_16x16x32_bf16(
                false, aq1.v, false, bk1.v, (short)0, s, false, false);
            bool msk = mask[(size_t)jc * BATCH + b] != 0;
#pragma unroll
            for (int r = 0; r < 8; ++r)
                p[jt][r] = msk ? -10000.0f : s[r] * 0.125f;   // 1/sqrt(64)
        }
        // ----- online softmax; xor 1/2/4/8 stays inside each 16-lane half ---
#pragma unroll
        for (int r = 0; r < 8; ++r) {
            float t = fmaxf(fmaxf(p[0][r], p[1][r]), fmaxf(p[2][r], p[3][r]));
            t = fmaxf(t, __shfl_xor(t, 1, 32));
            t = fmaxf(t, __shfl_xor(t, 2, 32));
            t = fmaxf(t, __shfl_xor(t, 4, 32));
            t = fmaxf(t, __shfl_xor(t, 8, 32));
            float nm = fmaxf(mrow[r], t);
            float alpha = __expf(mrow[r] - nm);
            mrow[r] = nm;
            float rs = 0.f;
#pragma unroll
            for (int jt = 0; jt < 4; ++jt) {
                p[jt][r] = __expf(p[jt][r] - nm);
                rs += p[jt][r];
            }
            rs += __shfl_xor(rs, 1, 32);
            rs += __shfl_xor(rs, 2, 32);
            rs += __shfl_xor(rs, 4, 32);
            rs += __shfl_xor(rs, 8, 32);
            lsum[r] = lsum[r] * alpha + rs;
#pragma unroll
            for (int t4 = 0; t4 < 4; ++t4) acc[t4][r] *= alpha;
        }
        // ----- P: C-layout -> LDS -> A-layout (same wave, DS in-order) ------
#pragma unroll
        for (int r = 0; r < 8; ++r) {
            int m = r + hi * 8;
#pragma unroll
            for (int jt = 0; jt < 4; ++jt)
                pbuf[m * 64 + jt * 16 + ln] = (__bf16)p[jt][r];
        }
        Frag16 ap0, ap1;
#pragma unroll
        for (int e = 0; e < 8; ++e) {
            ap0.v[e]     = pbuf[ln * 64 + kbA + e];
            ap0.v[e + 8] = pbuf[ln * 64 + 16 + kbA + e];
            ap1.v[e]     = pbuf[ln * 64 + 32 + kbA + e];
            ap1.v[e + 8] = pbuf[ln * 64 + 48 + kbA + e];
        }
        // ----- P(16x64) x V(64x64): V rows contiguous thanks to Vt layout ---
#pragma unroll
        for (int t4 = 0; t4 < 4; ++t4) {
            const __bf16* vtrow = Vt + ((size_t)bn * DHEAD + t4 * 16 + ln) * KVLEN;
            Frag16 bv0, bv1;
            bv0.q[0] = *(const uint4*)(vtrow + j0 + kbB);
            bv0.q[1] = *(const uint4*)(vtrow + j0 + kbB + 16);
            bv1.q[0] = *(const uint4*)(vtrow + j0 + 32 + kbB);
            bv1.q[1] = *(const uint4*)(vtrow + j0 + 32 + kbB + 16);
            acc[t4] = __builtin_amdgcn_wmma_f32_16x16x32_bf16(
                false, ap0.v, false, bv0.v, (short)0, acc[t4], false, false);
            acc[t4] = __builtin_amdgcn_wmma_f32_16x16x32_bf16(
                false, ap1.v, false, bv1.v, (short)0, acc[t4], false, false);
        }
    }

    // ----- normalize and store attn_vec [i][b][nh*64+d] as bf16 -------------
#pragma unroll
    for (int t4 = 0; t4 < 4; ++t4) {
#pragma unroll
        for (int r = 0; r < 8; ++r) {
            int i = qt * 16 + r + hi * 8;
            int d = t4 * 16 + ln;
            float val = acc[t4][r] / lsum[r];
            out[(size_t)i * (BATCH * DMODEL) + (size_t)b * DMODEL + nh * DHEAD + d] = (__bf16)val;
        }
    }
}

// ---------------- CDNA5 TDM probe (compiled, never launched) ------------------
// tensor_load_to_lds with an all-zero descriptor: D# group0.count==0 is the
// architecturally-defined NULL tensor (no-op).
__global__ void cdna5_probe_kernel(float* __restrict__ out)
{
#if __has_builtin(__builtin_amdgcn_tensor_load_to_lds)
    {
        u32x4 g0 = {}; i32x8 g1 = {}; i32x4 g2 = {}; i32x4 g3 = {};
#if defined(__clang_major__) && (__clang_major__ >= 23)
        i32x8 g4 = {};
        __builtin_amdgcn_tensor_load_to_lds(g0, g1, g2, g3, g4, 0);
#else
        __builtin_amdgcn_tensor_load_to_lds(g0, g1, g2, g3, 0);
#endif
    }
#endif
#if __has_builtin(__builtin_amdgcn_s_wait_tensorcnt)
    __builtin_amdgcn_s_wait_tensorcnt(0);
#endif
    out[threadIdx.x] = 0.f;
}

// ---------------- launcher ----------------
extern "C" void kernel_launch(void* const* d_in, const int* in_sizes, int n_in,
                              void* d_out, int out_size, void* d_ws, size_t ws_size,
                              hipStream_t stream)
{
    const float* h   = (const float*)d_in[0];
    const float* c   = (const float*)d_in[1];
    const unsigned char* mask = (const unsigned char*)d_in[2];  // jnp.bool_ = 1 byte
    const float* Wq  = (const float*)d_in[3];
    const float* Wkv = (const float*)d_in[4];
    const float* Wo  = (const float*)d_in[5];
    const float* gamma = (const float*)d_in[6];
    const float* beta  = (const float*)d_in[7];
    float* out = (float*)d_out;

    char* w = (char*)d_ws;
    float*  hlnF = (float*) (w + 0);            // 32 MiB
    __bf16* hlnB = (__bf16*)(w + 33554432);     // 16 MiB
    __bf16* cB   = (__bf16*)(w + 50331648);     // 16 MiB
    __bf16* WqT  = (__bf16*)(w + 67108864);     //  2 MiB  [N=1024 x K=1024]
    __bf16* WkvT = (__bf16*)(w + 69206016);     //  4 MiB  [N=2048 x K=1024]
    __bf16* WoT  = (__bf16*)(w + 73400320);     //  2 MiB  [N=1024 x K=1024]
    __bf16* qB   = (__bf16*)(w + 75497472);     // 16 MiB  [b][h][i][d]
    __bf16* kB   = (__bf16*)(w + 92274688);     // 16 MiB  [b][h][j][d]
    __bf16* vT   = (__bf16*)(w + 109051904);    // 16 MiB  [b][h][d][j]  (transposed)
    __bf16* atB  = (__bf16*)(w + 125829120);    // 16 MiB  [i][b][h*64+d]

    // 1) layernorm
    ln_kernel<<<QLEN * BATCH, 256, 0, stream>>>(h, gamma, beta, hlnF, hlnB);
    // 2) c -> bf16
    f2b_kernel<<<(KVLEN * BATCH * DMODEL) / 256, 256, 0, stream>>>(c, cB, KVLEN * BATCH * DMODEL);
    // 3) weights -> bf16 transposed [N x K]
    tconv_kernel<<<(DMODEL * DMODEL) / 256, 256, 0, stream>>>(Wq, WqT, DMODEL, DMODEL);
    tconv_kernel<<<(DMODEL * 2 * DMODEL) / 256, 256, 0, stream>>>(Wkv, WkvT, DMODEL, 2 * DMODEL);
    tconv_kernel<<<(DMODEL * DMODEL) / 256, 256, 0, stream>>>(Wo, WoT, DMODEL, DMODEL);
    // 4) Q = hln @ Wq  (M=8192,N=1024,K=1024): 256x16 tiles of 32x64 -> 512 blocks
    gemm_kernel<<<512, 256, 0, stream>>>(hlnB, WqT, QLEN * BATCH, DMODEL, DMODEL, 0,
                                         nullptr, nullptr, qB, nullptr, nullptr);
    // 5) KV = c @ Wkv (M=8192,N=2048,K=1024): 256x32 tiles -> 1024 blocks
    gemm_kernel<<<1024, 256, 0, stream>>>(cB, WkvT, KVLEN * BATCH, 2 * DMODEL, DMODEL, 1,
                                          nullptr, nullptr, nullptr, kB, vT);
    // 6) flash attention (8192 waves -> 1024 blocks)
    attn_kernel<<<1024, 256, 0, stream>>>(qB, kB, vT, mask, atB);
    // 7) out = hln + attn_vec @ Wo
    gemm_kernel<<<512, 256, 0, stream>>>(atB, WoT, QLEN * BATCH, DMODEL, DMODEL, 2,
                                         hlnF, out, nullptr, nullptr, nullptr);
}
